// Stage2_42322607735561
// MI455X (gfx1250) — matrix operand order repaired and checked by reading, not compiled, not run on hardware
//
#include <hip/hip_runtime.h>
#include <hip/hip_bf16.h>

#define T_STEPS 256
#define BATCH   64
#define HID     1024
#define GCOLS   4096   // 4*HID, gate-interleaved: col' = 4*j + gate

typedef __attribute__((ext_vector_type(16))) _Float16 v16h;
typedef __attribute__((ext_vector_type(8)))  _Float16 v8h;
typedef __attribute__((ext_vector_type(8)))  float    v8f;
typedef __attribute__((ext_vector_type(4)))  float    v4f;

union HFrag { v16h v; v8h h[2]; };

static __device__ __forceinline__ v16h make_frag(v8h lo, v8h hi) {
    HFrag f; f.h[0] = lo; f.h[1] = hi; return f.v;
}
static __device__ __forceinline__ float sigmoidf_(float x) {
    return 1.0f / (1.0f + __expf(-x));
}

// ---------------------------------------------------------------------------
// CDNA5 async copy global -> LDS (ASYNCcnt-tracked, no VGPR round trip).
// INST_OFFSET is added to BOTH the LDS and global addresses (ISA 15.18.3).
// ---------------------------------------------------------------------------
template <int OFF>
static __device__ __forceinline__ void async_copy_b128(unsigned lds_off, const _Float16* g) {
    asm volatile("global_load_async_to_lds_b128 %0, %1, off offset:%c2"
                 :: "v"(lds_off), "v"(g), "i"(OFF)
                 : "memory");
}
static __device__ __forceinline__ void wait_async_0() { asm volatile("s_wait_asynccnt 0x0" ::: "memory"); }
static __device__ __forceinline__ void wait_async_1() { asm volatile("s_wait_asynccnt 0x1" ::: "memory"); }
static __device__ __forceinline__ void wait_async_2() { asm volatile("s_wait_asynccnt 0x2" ::: "memory"); }
static __device__ __forceinline__ void wait_async_4() { asm volatile("s_wait_asynccnt 0x4" ::: "memory"); }

// ---------------------------------------------------------------------------
// Weight pack: f32 (4H,H) row-major -> f16 with gate-interleaved rows.
// out row g' = 4*j + gate  <-  in row gate*H + j.  (k stays contiguous, so
// WMMA B-fragments are two 16B loads per lane straight from this buffer.)
// ---------------------------------------------------------------------------
__global__ void pack_w_kernel(const float* __restrict__ w, _Float16* __restrict__ wp) {
    int o  = blockIdx.x * 256 + threadIdx.x;      // 0 .. 4096*1024-1
    int gp = o >> 10;
    int k  = o & 1023;
    int j = gp >> 2, gate = gp & 3;
    wp[o] = (_Float16)w[(size_t)(gate * HID + j) * HID + k];
}

__global__ void pack_bias_kernel(const float* __restrict__ bih,
                                 const float* __restrict__ bhh,
                                 float* __restrict__ bp) {
    int o = blockIdx.x * 256 + threadIdx.x;       // 0..4095
    int j = o >> 2, gate = o & 3;
    bp[o] = bih[gate * HID + j] + bhh[gate * HID + j];
}

__global__ void f32_to_f16_kernel(const float* __restrict__ in,
                                  _Float16* __restrict__ out, int n) {
    int i = blockIdx.x * 256 + threadIdx.x;
    if (i < n) out[i] = (_Float16)in[i];
}

__global__ void init_kernel(_Float16* __restrict__ hbuf, unsigned* __restrict__ cnt) {
    int i = blockIdx.x * 256 + threadIdx.x;
    if (i == 0) *cnt = 0u;
    if (i < 2 * BATCH * HID) hbuf[i] = (_Float16)0.0f;
}

// ---------------------------------------------------------------------------
// Input GEMM: XG[16384][4096] = A_f16[16384][1024] x Wp_f16[4096][1024]^T + b
// WG = 256 threads (8 waves), tile 128(M) x 64(N); wave: 1 N-tile, 4 M-tiles.
// A chunks async-staged into a 4-deep LDS ring (3-stage lookahead, one
// barrier per chunk). B fragments prefetched one chunk ahead from global.
// ---------------------------------------------------------------------------
__global__ __launch_bounds__(256) void input_gemm_kernel(
    const _Float16* __restrict__ A,
    const _Float16* __restrict__ Wp,
    const float*    __restrict__ bias,
    float*          __restrict__ XG)
{
    __shared__ _Float16 lds_a[4][128 * 32];

    int bx      = blockIdx.x;
    int mblk    = bx & 127;
    int nblk    = bx >> 7;
    int rowbase = mblk * 128;
    int colbase = nblk * 64;
    int tid  = threadIdx.x;
    int lane = tid & 31, wave = tid >> 5;
    int nt   = wave & 3;
    int mtb  = (wave >> 2) * 4;
    int lcol = lane & 15, lhalf = lane >> 4;

    // staging: each thread copies 32B (two async b128) per chunk
    int srow  = tid >> 1;
    int skoff = (tid & 1) * 16;
    const _Float16* ag = A + (size_t)(rowbase + srow) * HID + skoff;
    unsigned ldsoff[4];
#pragma unroll
    for (int bfi = 0; bfi < 4; ++bfi)
        ldsoff[bfi] = (unsigned)(uintptr_t)(&lds_a[bfi][srow * 32 + skoff]);

    auto stage = [&](int kc) {
        const _Float16* src = ag + kc * 32;
        unsigned dst = ldsoff[kc & 3];
        async_copy_b128<0>(dst, src);
        async_copy_b128<16>(dst, src);
    };

    stage(0); stage(1); stage(2);

    const _Float16* bbase = Wp + (size_t)(colbase + nt * 16 + lcol) * HID + lhalf * 8;
    v16h bf = make_frag(*(const v8h*)(bbase), *(const v8h*)(bbase + 16));

    v8f acc[4] = {};
    for (int kc = 0; kc < 32; ++kc) {
        if (kc < 30)       wait_async_4();   // 3 stages in flight -> oldest done
        else if (kc == 30) wait_async_2();
        else               wait_async_0();
        __syncthreads();

        v16h bfn = bf;
        if (kc < 31) {
            const _Float16* bp = bbase + (kc + 1) * 32;
            bfn = make_frag(*(const v8h*)bp, *(const v8h*)(bp + 16));
        }
        const _Float16* lbuf = lds_a[kc & 3];
#pragma unroll
        for (int m = 0; m < 4; ++m) {
            const _Float16* ap = lbuf + ((mtb + m) * 16 + lcol) * 32 + lhalf * 8;
            v16h af = make_frag(*(const v8h*)ap, *(const v8h*)(ap + 16));
            acc[m] = __builtin_amdgcn_wmma_f32_16x16x32_f16(
                false, af, false, bf, (short)0, acc[m], false, false);
        }
        if (kc + 3 < 32) stage(kc + 3);   // writes buf (kc+3)&3, disjoint from readers
        bf = bfn;
    }

    int   col = colbase + nt * 16 + lcol;
    float bv  = bias[col];
#pragma unroll
    for (int m = 0; m < 4; ++m) {
        int r0 = rowbase + (mtb + m) * 16 + lhalf * 8;
#pragma unroll
        for (int v = 0; v < 8; ++v)
            XG[(size_t)(r0 + v) * GCOLS + col] = acc[m][v] + bv;
    }
}

// ---------------------------------------------------------------------------
// Persistent LSTM scan. 64 WGs x 256 threads; WG 'wg' owns gate columns
// [wg*64, wg*64+64) == h columns [wg*16, wg*16+16) with all 4 gates local.
// Per step: gates = h_{t-1} @ Whh^T (WMMA, K=1024, async-staged h), + XG[t]
// (prefetched to registers before the K loop), cell update with c in
// registers, f16 h write (double buffered), device-wide barrier.
// ---------------------------------------------------------------------------
__global__ __launch_bounds__(256) void lstm_scan_kernel(
    const float*    __restrict__ XG,
    const _Float16* __restrict__ Whh,
    const float*    __restrict__ resid,
    float*          __restrict__ out,
    _Float16*       __restrict__ out_f16,   // may be null
    _Float16*       __restrict__ hbuf,      // [2][B][H] f16, zeroed
    unsigned*       __restrict__ cnt,
    int write_f16)
{
    __shared__ _Float16 lds_h[4][BATCH * 32];
    __shared__ float    lds_g[BATCH * 64];

    int wg      = blockIdx.x;
    int colbase = wg * 64;
    int tid  = threadIdx.x;
    int lane = tid & 31, wave = tid >> 5;
    int nt   = wave >> 1;
    int mtb  = (wave & 1) * 2;
    int lcol = lane & 15, lhalf = lane >> 4;

    const _Float16* bbase = Whh + (size_t)(colbase + nt * 16 + lcol) * HID + lhalf * 8;

    int srow  = tid >> 2;             // 0..63
    int skoff = (tid & 3) * 8;        // one async b128 per thread per chunk
    unsigned ldsoff[4];
#pragma unroll
    for (int bfi = 0; bfi < 4; ++bfi)
        ldsoff[bfi] = (unsigned)(uintptr_t)(&lds_h[bfi][srow * 32 + skoff]);

    float cstate[4] = {0.f, 0.f, 0.f, 0.f};
    unsigned nwg = gridDim.x;

    for (int step = 0; step < T_STEPS; ++step) {
        const _Float16* hread  = hbuf + (size_t)(step & 1) * BATCH * HID;
        _Float16*       hwrite = hbuf + (size_t)((step + 1) & 1) * BATCH * HID;

        // prefetch this step's gate-input slice and residuals (independent
        // of h): latency hides behind the whole recurrent GEMM below.
        v4f   xg4[4];
        float rs4[4];
#pragma unroll
        for (int q = 0; q < 4; ++q) {
            int idx = tid + 256 * q;
            int b   = idx >> 4;
            int j   = idx & 15;
            xg4[q] = *(const v4f*)(XG + (size_t)(step * BATCH + b) * GCOLS + colbase + j * 4);
            rs4[q] = resid[(size_t)(step * BATCH + b) * HID + wg * 16 + j];
        }

        auto stage = [&](int kc) {
            async_copy_b128<0>(ldsoff[kc & 3], hread + (size_t)srow * HID + kc * 32 + skoff);
        };
        stage(0); stage(1); stage(2);

        v16h bf = make_frag(*(const v8h*)(bbase), *(const v8h*)(bbase + 16));
        v8f acc[2] = {};
        for (int kc = 0; kc < 32; ++kc) {
            if (kc < 30)       wait_async_2();
            else if (kc == 30) wait_async_1();
            else               wait_async_0();
            __syncthreads();

            v16h bfn = bf;
            if (kc < 31) {
                const _Float16* bp = bbase + (kc + 1) * 32;
                bfn = make_frag(*(const v8h*)bp, *(const v8h*)(bp + 16));
            }
            const _Float16* lbuf = lds_h[kc & 3];
#pragma unroll
            for (int m = 0; m < 2; ++m) {
                const _Float16* ap = lbuf + ((mtb + m) * 16 + lcol) * 32 + lhalf * 8;
                v16h af = make_frag(*(const v8h*)ap, *(const v8h*)(ap + 16));
                acc[m] = __builtin_amdgcn_wmma_f32_16x16x32_f16(
                    false, af, false, bf, (short)0, acc[m], false, false);
            }
            if (kc + 3 < 32) stage(kc + 3);
            bf = bfn;
        }

        // gate accumulators -> LDS [64 rows][64 cols]
#pragma unroll
        for (int m = 0; m < 2; ++m) {
            int r0 = (mtb + m) * 16 + lhalf * 8;
#pragma unroll
            for (int v = 0; v < 8; ++v)
                lds_g[(r0 + v) * 64 + nt * 16 + lcol] = acc[m][v];
        }
        __syncthreads();

        // cell update: thread handles 4 (b, j) cells; i,f,g,o interleaved.
#pragma unroll
        for (int q = 0; q < 4; ++q) {
            int idx = tid + 256 * q;
            int b   = idx >> 4;
            int j   = idx & 15;
            v4f gv = *(const v4f*)(lds_g + b * 64 + j * 4);
            float gi = sigmoidf_(gv[0] + xg4[q][0]);
            float gf = sigmoidf_(gv[1] + xg4[q][1]);
            float gg = tanhf    (gv[2] + xg4[q][2]);
            float go = sigmoidf_(gv[3] + xg4[q][3]);
            float c  = gf * cstate[q] + gi * gg;
            cstate[q] = c;
            float h  = go * tanhf(c);
            int jg = wg * 16 + j;
            size_t ridx = (size_t)(step * BATCH + b) * HID + jg;
            float ov = h + rs4[q];
            out[ridx] = ov;
            hwrite[b * HID + jg] = (_Float16)h;
            if (write_f16) out_f16[ridx] = (_Float16)ov;
        }
        __syncthreads();

        // device-wide barrier: all h_t writes visible before step t+1 reads
        __threadfence();
        if (tid == 0) {
            __hip_atomic_fetch_add(cnt, 1u, __ATOMIC_ACQ_REL, __HIP_MEMORY_SCOPE_AGENT);
            unsigned target = (unsigned)(step + 1) * nwg;
            while (__hip_atomic_load(cnt, __ATOMIC_ACQUIRE, __HIP_MEMORY_SCOPE_AGENT) < target)
                __builtin_amdgcn_s_sleep(2);
        }
        __syncthreads();
    }
}

// ---------------------------------------------------------------------------
extern "C" void kernel_launch(void* const* d_in, const int* in_sizes, int n_in,
                              void* d_out, int out_size, void* d_ws, size_t ws_size,
                              hipStream_t stream)
{
    const float* x    = (const float*)d_in[0];
    const float* wih5 = (const float*)d_in[1];
    const float* whh5 = (const float*)d_in[2];
    const float* bih5 = (const float*)d_in[3];
    const float* bhh5 = (const float*)d_in[4];
    const float* wih6 = (const float*)d_in[5];
    const float* whh6 = (const float*)d_in[6];
    const float* bih6 = (const float*)d_in[7];
    const float* bhh6 = (const float*)d_in[8];
    float* out6 = (float*)d_out;

    char* wsb = (char*)d_ws;
    size_t off = 0;
    auto wsalloc = [&](size_t bytes) -> void* {
        void* p = wsb + off;
        off += (bytes + 255) & ~(size_t)255;
        return p;
    };
    const size_t M = (size_t)T_STEPS * BATCH;                 // 16384
    float*    XG    = (float*)   wsalloc(M * GCOLS * 4);
    _Float16* XF16  = (_Float16*)wsalloc(M * HID * 2);
    float*    O5    = (float*)   wsalloc(M * HID * 4);
    _Float16* O5F16 = (_Float16*)wsalloc(M * HID * 2);
    _Float16* WIH5P = (_Float16*)wsalloc((size_t)GCOLS * HID * 2);
    _Float16* WHH5P = (_Float16*)wsalloc((size_t)GCOLS * HID * 2);
    _Float16* WIH6P = (_Float16*)wsalloc((size_t)GCOLS * HID * 2);
    _Float16* WHH6P = (_Float16*)wsalloc((size_t)GCOLS * HID * 2);
    float*    B5    = (float*)   wsalloc(GCOLS * 4);
    float*    B6    = (float*)   wsalloc(GCOLS * 4);
    _Float16* HBUF  = (_Float16*)wsalloc((size_t)2 * BATCH * HID * 2);
    unsigned* CNT   = (unsigned*)wsalloc(256);

    pack_w_kernel<<<16384, 256, 0, stream>>>(wih5, WIH5P);
    pack_w_kernel<<<16384, 256, 0, stream>>>(whh5, WHH5P);
    pack_w_kernel<<<16384, 256, 0, stream>>>(wih6, WIH6P);
    pack_w_kernel<<<16384, 256, 0, stream>>>(whh6, WHH6P);
    pack_bias_kernel<<<16, 256, 0, stream>>>(bih5, bhh5, B5);
    pack_bias_kernel<<<16, 256, 0, stream>>>(bih6, bhh6, B6);
    f32_to_f16_kernel<<<(int)((M * HID + 255) / 256), 256, 0, stream>>>(x, XF16, (int)(M * HID));

    // layer 5
    input_gemm_kernel<<<8192, 256, 0, stream>>>(XF16, WIH5P, B5, XG);
    init_kernel<<<512, 256, 0, stream>>>(HBUF, CNT);
    lstm_scan_kernel<<<64, 256, 0, stream>>>(XG, WHH5P, x, O5, O5F16, HBUF, CNT, 1);

    // layer 6 (reuses XG)
    input_gemm_kernel<<<8192, 256, 0, stream>>>(O5F16, WIH6P, B6, XG);
    init_kernel<<<512, 256, 0, stream>>>(HBUF, CNT);
    lstm_scan_kernel<<<64, 256, 0, stream>>>(XG, WHH6P, O5, out6, (_Float16*)nullptr, HBUF, CNT, 0);
}
